// LinearMultiHeadAttention_5866925326851
// MI455X (gfx1250) — matrix operand order repaired and checked by reading, not compiled, not run on hardware
//
#include <hip/hip_runtime.h>
#include <math.h>

#define DIN  8448
#define NTOK 8192
#define DOUT 256
#define NH   4
#define HD   64

typedef __attribute__((ext_vector_type(16))) __bf16          v16bf;
typedef __attribute__((ext_vector_type(16))) unsigned short  v16us;
typedef __attribute__((ext_vector_type(8)))  float           v8f;

// Split an fp32 into bf16 hi + bf16 residual (truncation split; hi+lo carries
// ~16 mantissa bits, product-sum error ~1e-5 relative over K=8448).
static __device__ __forceinline__ void split2(float f, unsigned short& hi, unsigned short& lo) {
    unsigned u = __float_as_uint(f);
    unsigned short h = (unsigned short)(u >> 16);
    float fh = __uint_as_float(((unsigned)h) << 16);
    float r  = f - fh;
    hi = h;
    lo = (unsigned short)(__float_as_uint(r) >> 16);
}

// acc += Ahi*Bhi + Ahi*Blo + Alo*Bhi   (fp32 accumulate, 3x v_wmma_f32_16x16x32_bf16)
static __device__ __forceinline__ v8f wmma3(v16us ah, v16us al, v16us bh, v16us bl, v8f c) {
    v16bf A  = __builtin_bit_cast(v16bf, ah);
    v16bf Al = __builtin_bit_cast(v16bf, al);
    v16bf B  = __builtin_bit_cast(v16bf, bh);
    v16bf Bl = __builtin_bit_cast(v16bf, bl);
    c = __builtin_amdgcn_wmma_f32_16x16x32_bf16(false, A,  false, B,  (short)0, c, false, false);
    c = __builtin_amdgcn_wmma_f32_16x16x32_bf16(false, A,  false, Bl, (short)0, c, false, false);
    c = __builtin_amdgcn_wmma_f32_16x16x32_bf16(false, Al, false, B,  (short)0, c, false, false);
    return c;
}

// ---------------------------------------------------------------------------
// Kernel 0: W1 [256, 8448] fp32 -> bf16 hi/lo (K-major rows, 32B aligned)
// ---------------------------------------------------------------------------
__global__ void lmha_wsplit(const float* __restrict__ W,
                            unsigned short* __restrict__ Whi,
                            unsigned short* __restrict__ Wlo) {
    size_t i = (size_t)blockIdx.x * blockDim.x + threadIdx.x; // grid sized exactly
    unsigned short h, l;
    split2(W[i], h, l);
    Whi[i] = h;
    Wlo[i] = l;
}

// ---------------------------------------------------------------------------
// Kernel 1: all three projections in one launch (grid = (N/16, 3)).
// One wave computes a 16-row x 256-col strip; X rows stream from HBM once,
// W hi/lo stream from L2 (8.6 MB, L2-resident).
// A-frag (16-bit A 16x32): lane m = l15, 16 contiguous K at kh*16.
// B-frag (16-bit B 32x16): lane n = l15, 16 contiguous K at kh*16.
// W tile offsets across nt are compile-time constants -> folded into the
// global_load_b128 24-bit instruction offset. B fragments are double-buffered
// one tile ahead so each tile's L2 latency overlaps the previous tile's WMMAs.
// ---------------------------------------------------------------------------
__global__ __launch_bounds__(32) void lmha_proj3(const float* __restrict__ Xq,
                                                 const float* __restrict__ Xk,
                                                 const float* __restrict__ Xv,
                                                 const unsigned short* __restrict__ Whi,
                                                 const unsigned short* __restrict__ Wlo,
                                                 const float* __restrict__ bias,
                                                 float* __restrict__ Yq,
                                                 float* __restrict__ Yk,
                                                 float* __restrict__ Yv) {
    const int lane = threadIdx.x;
    const int l15  = lane & 15;
    const int kh   = lane >> 4;
    const int row0 = blockIdx.x * 16;
    const int m    = blockIdx.y;                 // 0=q, 1=k, 2=v
    const float* X = (m == 0) ? Xq : (m == 1) ? Xk : Xv;
    float*       Y = (m == 0) ? Yq : (m == 1) ? Yk : Yv;
    const int feat = (m != 2);

    v8f acc[16];
#pragma unroll
    for (int nt = 0; nt < 16; ++nt) {
        float bv = bias[nt * 16 + l15];          // bias depends on N only
#pragma unroll
        for (int j = 0; j < 8; ++j) acc[nt][j] = bv;
    }

    // Per-lane streaming pointers; bumped by 32 elements per K-step so all
    // tile addressing reduces to immediate offsets.
    const float*          xp  = X   + (size_t)(row0 + l15) * DIN + kh * 16;
    const unsigned short* whp = Whi + (size_t)l15 * DIN + kh * 16;
    const unsigned short* wlp = Wlo + (size_t)l15 * DIN + kh * 16;

    // Preload B tile 0 of the first K-step.
    v16us bh_cur = *(const v16us*)whp;
    v16us bl_cur = *(const v16us*)wlp;

    for (int k0 = 0; k0 < DIN; k0 += 32) {
        float4 a0 = ((const float4*)xp)[0];
        float4 a1 = ((const float4*)xp)[1];
        float4 a2 = ((const float4*)(xp + 8))[0];
        float4 a3 = ((const float4*)(xp + 8))[1];
        float fx[16] = {a0.x, a0.y, a0.z, a0.w, a1.x, a1.y, a1.z, a1.w,
                        a2.x, a2.y, a2.z, a2.w, a3.x, a3.y, a3.z, a3.w};
        v16us ah, al;
#pragma unroll
        for (int e = 0; e < 16; ++e) {
            unsigned short h, l;
            split2(fx[e], h, l);
            ah[e] = h; al[e] = l;
        }
#pragma unroll
        for (int nt = 0; nt < 16; ++nt) {
            // Issue next tile's loads before consuming the current tile.
            // At nt==15, preload tile 0 of the next K-step (the final dangling
            // preload past the last K-step stays inside the workspace and its
            // value is never used).
            const size_t noff = (nt < 15) ? (size_t)(nt + 1) * 16 * DIN : (size_t)32;
            v16us bh_nxt = *(const v16us*)(whp + noff);
            v16us bl_nxt = *(const v16us*)(wlp + noff);
            acc[nt] = wmma3(ah, al, bh_cur, bl_cur, acc[nt]);
            bh_cur = bh_nxt;
            bl_cur = bl_nxt;
        }
        xp  += 32;
        whp += 32;
        wlp += 32;
    }

    // D layout: vgpr r -> row (r + kh*8), col = nt*16 + l15
#pragma unroll
    for (int nt = 0; nt < 16; ++nt) {
        const int n = nt * 16 + l15;
#pragma unroll
        for (int r = 0; r < 8; ++r) {
            float v = acc[nt][r];
            if (feat) v = (v > 0.f) ? (v + 1.f) : expf(v);   // elu(x)+1
            Y[(size_t)(row0 + r + kh * 8) * DOUT + n] = v;
        }
    }
}

// ---------------------------------------------------------------------------
// Kernel 2: partial key_values[h,kv,d] = sum_t v[t,h,kv] * k[t,h,d]
// per head per K-slice of 1024 tokens (deterministic two-stage reduction).
// ---------------------------------------------------------------------------
__global__ __launch_bounds__(32) void lmha_kvpart(const float* __restrict__ kf,
                                                  const float* __restrict__ vf,
                                                  float* __restrict__ pkv) {
    const int lane = threadIdx.x;
    const int l15  = lane & 15;
    const int kh   = lane >> 4;
    const int h     = blockIdx.y;
    const int slice = blockIdx.x;
    const int tb    = slice * (NTOK / 8);

    v8f acc[16];
#pragma unroll
    for (int i = 0; i < 16; ++i)
#pragma unroll
        for (int j = 0; j < 8; ++j) acc[i][j] = 0.f;

    for (int t0 = tb; t0 < tb + NTOK / 8; t0 += 32) {
        v16us ah[4], al[4];
#pragma unroll
        for (int mt = 0; mt < 4; ++mt) {
            const int kv = mt * 16 + l15;
#pragma unroll
            for (int e = 0; e < 16; ++e) {
                const int t = t0 + kh * 16 + e;
                unsigned short hh, ll;
                split2(vf[(size_t)t * DOUT + h * HD + kv], hh, ll);
                ah[mt][e] = hh; al[mt][e] = ll;
            }
        }
#pragma unroll
        for (int nt = 0; nt < 4; ++nt) {
            const int d  = nt * 16 + l15;
            const int ts = t0 + kh * 16;
            v16us bh, bl;
#pragma unroll
            for (int e = 0; e < 16; ++e) {
                unsigned short hh, ll;
                split2(kf[(size_t)(ts + e) * DOUT + h * HD + d], hh, ll);
                bh[e] = hh; bl[e] = ll;
            }
#pragma unroll
            for (int mt = 0; mt < 4; ++mt)
                acc[mt * 4 + nt] = wmma3(ah[mt], al[mt], bh, bl, acc[mt * 4 + nt]);
        }
    }

    float* base = pkv + (size_t)(slice * NH + h) * (HD * HD);
#pragma unroll
    for (int mt = 0; mt < 4; ++mt)
#pragma unroll
        for (int nt = 0; nt < 4; ++nt)
#pragma unroll
            for (int r = 0; r < 8; ++r) {
                const int kv = mt * 16 + r + kh * 8;
                const int d  = nt * 16 + l15;
                base[kv * HD + d] = acc[mt * 4 + nt][r];
            }
}

__global__ void lmha_kvreduce(const float* __restrict__ pkv, float* __restrict__ kvm) {
    const int i = blockIdx.x * 256 + threadIdx.x;   // < NH*HD*HD = 16384
    float s = 0.f;
#pragma unroll
    for (int sl = 0; sl < 8; ++sl) s += pkv[(size_t)sl * NH * HD * HD + i];
    kvm[i] = s;
}

// k_sum: partial column sums over 256-row chunks, then reduce.
__global__ void lmha_kspart(const float* __restrict__ kf, float* __restrict__ pks) {
    const int col = threadIdx.x;
    const int r0  = blockIdx.x * (NTOK / 32);
    float s = 0.f;
    for (int r = r0; r < r0 + NTOK / 32; ++r) s += kf[(size_t)r * DOUT + col];
    pks[blockIdx.x * DOUT + col] = s;
}

__global__ void lmha_ksreduce(const float* __restrict__ pks, float* __restrict__ ks) {
    const int col = threadIdx.x;
    float s = 0.f;
#pragma unroll
    for (int b = 0; b < 32; ++b) s += pks[b * DOUT + col];
    ks[col] = s;
}

// ---------------------------------------------------------------------------
// Kernel 3: out[t, h*64+kv] = (sum_d q[t,h,d]*KV[h,kv,d]) / (q[t,h,:].k_sum[h] + eps)
// One wave per (16 tokens, head). WMMA over K=64 (2 steps) x 4 N-tiles.
// ---------------------------------------------------------------------------
__global__ __launch_bounds__(32) void lmha_attn(const float* __restrict__ qf,
                                                const float* __restrict__ kvm,
                                                const float* __restrict__ ks,
                                                float* __restrict__ out) {
    const int lane = threadIdx.x;
    const int l15  = lane & 15;
    const int kh   = lane >> 4;
    const int t0   = blockIdx.x * 16;
    const int h    = blockIdx.y;

    const float* qrow = qf + (size_t)(t0 + l15) * DOUT + h * HD;
    float dot = 0.f;
#pragma unroll
    for (int d = 0; d < HD; ++d) dot += qrow[d] * ks[h * HD + d];
    const float nrm = 1.f / (dot + 1e-6f);

    v8f acc[4];
#pragma unroll
    for (int i = 0; i < 4; ++i)
#pragma unroll
        for (int j = 0; j < 8; ++j) acc[i][j] = 0.f;

#pragma unroll
    for (int kst = 0; kst < 2; ++kst) {
        const int base = kst * 32;
        v16us ah, al;
#pragma unroll
        for (int e = 0; e < 16; ++e) {
            const int c = base + kh * 16 + e;
            unsigned short hh, ll;
            split2(qrow[c], hh, ll);
            ah[e] = hh; al[e] = ll;
        }
#pragma unroll
        for (int nt = 0; nt < 4; ++nt) {
            const int kvcol = nt * 16 + l15;
            const float* bp = kvm + (size_t)h * HD * HD + (size_t)kvcol * HD + base + kh * 16;
            v16us bh, bl;
#pragma unroll
            for (int e = 0; e < 16; ++e) {
                unsigned short hh, ll;
                split2(bp[e], hh, ll);
                bh[e] = hh; bl[e] = ll;
            }
            acc[nt] = wmma3(ah, al, bh, bl, acc[nt]);
        }
    }

#pragma unroll
    for (int nt = 0; nt < 4; ++nt)
#pragma unroll
        for (int r = 0; r < 8; ++r) {
            const int row = r + kh * 8;
            const float nv = __shfl(nrm, row, 32);   // norm lives on lanes 0..15
            out[(size_t)(t0 + row) * DOUT + h * HD + nt * 16 + l15] = acc[nt][r] * nv;
        }
}

// ---------------------------------------------------------------------------
extern "C" void kernel_launch(void* const* d_in, const int* in_sizes, int n_in,
                              void* d_out, int out_size, void* d_ws, size_t ws_size,
                              hipStream_t stream) {
    const float* Q  = (const float*)d_in[0];
    const float* K  = (const float*)d_in[1];
    const float* V  = (const float*)d_in[2];
    const float* W1 = (const float*)d_in[3];
    const float* b1 = (const float*)d_in[4];
    float* out = (float*)d_out;

    char* w = (char*)d_ws;
    size_t off = 0;
    auto take = [&](size_t bytes) -> char* {
        char* p = w + off;
        off += (bytes + 255) & ~(size_t)255;
        return p;
    };
    unsigned short* Whi = (unsigned short*)take((size_t)DOUT * DIN * 2);
    unsigned short* Wlo = (unsigned short*)take((size_t)DOUT * DIN * 2);
    float* qf  = (float*)take((size_t)NTOK * DOUT * 4);
    float* kf  = (float*)take((size_t)NTOK * DOUT * 4);
    float* vf  = (float*)take((size_t)NTOK * DOUT * 4);
    float* pkv = (float*)take((size_t)8 * NH * HD * HD * 4);
    float* kvm = (float*)take((size_t)NH * HD * HD * 4);
    float* pks = (float*)take((size_t)32 * DOUT * 4);
    float* ks  = (float*)take((size_t)DOUT * 4);

    lmha_wsplit<<<(DOUT * DIN) / 256, 256, 0, stream>>>(W1, Whi, Wlo);

    lmha_proj3<<<dim3(NTOK / 16, 3), 32, 0, stream>>>(Q, K, V, Whi, Wlo, b1, qf, kf, vf);

    lmha_kvpart<<<dim3(8, NH), 32, 0, stream>>>(kf, vf, pkv);
    lmha_kspart<<<32, DOUT, 0, stream>>>(kf, pks);
    lmha_kvreduce<<<(NH * HD * HD) / 256, 256, 0, stream>>>(pkv, kvm);
    lmha_ksreduce<<<1, DOUT, 0, stream>>>(pks, ks);

    lmha_attn<<<dim3(NTOK / 16, NH), 32, 0, stream>>>(qf, kvm, ks, out);
}
